// MPNEncoder_64476049047871
// MI455X (gfx1250) — compile-verified, matrix operand order checked by reading, not run on hardware
//
#include <hip/hip_runtime.h>
#include <hip/hip_bf16.h>

// ---------------------------------------------------------------------------
// Types for CDNA5 WMMA (gfx1250): v_wmma_f32_16x16x32_bf16
// ---------------------------------------------------------------------------
typedef __attribute__((ext_vector_type(16))) __bf16    v16bf;
typedef __attribute__((ext_vector_type(8)))  float     v8f;
typedef __attribute__((ext_vector_type(8)))  unsigned  v8u;

// Round-to-nearest-even pack of two f32 into one u32 holding two bf16.
__device__ __forceinline__ unsigned pack2bf(float lo, float hi) {
  unsigned ul = __builtin_bit_cast(unsigned, lo);
  unsigned uh = __builtin_bit_cast(unsigned, hi);
  ul += 0x7fffu + ((ul >> 16) & 1u);
  uh += 0x7fffu + ((uh >> 16) & 1u);
  return (ul >> 16) | (uh & 0xffff0000u);
}

// ---------------------------------------------------------------------------
// f32 [rows,K] -> bf16 [rows,Kp] (Kp%32==0, pad zeroed). One u32 (2 bf16) per
// thread. If rows_div != nullptr, effective rows = rows / *rows_div.
// ---------------------------------------------------------------------------
__global__ void to_bf16(const float* __restrict__ src, unsigned* __restrict__ dst,
                        int rows, int K, int Kp, const int* __restrict__ rows_div) {
  if (rows_div) rows = rows / *rows_div;
  long long t = (long long)blockIdx.x * blockDim.x + threadIdx.x;
  const int pairs = Kp >> 1;
  if (t >= (long long)rows * pairs) return;
  int row = (int)(t / pairs);
  int k = (int)(t - (long long)row * pairs) * 2;
  const float* s = src + (size_t)row * K;
  float f0 = (k < K)     ? s[k]     : 0.f;
  float f1 = (k + 1 < K) ? s[k + 1] : 0.f;
  dst[(size_t)row * (Kp >> 1) + (k >> 1)] = pack2bf(f0, f1);
}

// ---------------------------------------------------------------------------
// Fragment loader: bf16 source, K%32==0, rows 16B-aligned.
// Lane layout per CDNA5 ISA 7.12.2 (16-bit A/B matrix, 16x32):
// lane = (half<<4)|r16; u32 slot j<4 -> K = k0 + half*8 + 2j (contig 8 vals),
// slot j>=4 -> K = k0 + 16 + half*8 + 2(j-4) (second contig 8-val run).
// Two contiguous 8xbf16 runs -> two global_load_b128, zero conversion ALU.
// ---------------------------------------------------------------------------
__device__ __forceinline__ v8u frag_bf16(const unsigned short* __restrict__ row,
                                         int k0, int half) {
  const uint4* p = (const uint4*)(row + k0 + half * 8);
  const uint4* q = (const uint4*)(row + k0 + 16 + half * 8);
  uint4 a = *p, b = *q;
  v8u r;
  r[0] = a.x; r[1] = a.y; r[2] = a.z; r[3] = a.w;
  r[4] = b.x; r[5] = b.y; r[6] = b.z; r[7] = b.w;
  return r;
}

#define WMMA_BF16(a, b, c)                                              \
  __builtin_amdgcn_wmma_f32_16x16x32_bf16(                              \
      false, __builtin_bit_cast(v16bf, a), false,                       \
      __builtin_bit_cast(v16bf, b), (short)0, (c), false, false)

// ---------------------------------------------------------------------------
// WMMA GEMM:  C[M,N] = act( res + A[M,K] * W[N,K]^T + bias ), bf16 operands.
// K must be a multiple of 32 (pre-padded). Block = 128 threads (4 waves);
// wave computes a 16x64 tile. Grid = (ceil(M/64), N/64), N%64==0.
// If Mdiv != nullptr, effective M = M / *Mdiv.
// ---------------------------------------------------------------------------
template <bool BIAS, bool RELU, bool RES>
__global__ __launch_bounds__(128) void gemm_wmma(
    const unsigned short* __restrict__ A, const unsigned short* __restrict__ W,
    const float* __restrict__ bias, const float* __restrict__ res,
    float* __restrict__ C, int M, int N, int K, const int* __restrict__ Mdiv) {
  if (Mdiv) M = M / *Mdiv;
  const int lane = threadIdx.x & 31;
  const int wave = threadIdx.x >> 5;
  const int half = lane >> 4;
  const int l16  = lane & 15;
  const int m0 = (blockIdx.x * 4 + wave) * 16;   // wave-uniform
  const int n0 = blockIdx.y * 64;
  if (m0 >= M) return;                           // uniform per wave -> EXEC ok

  // Clamp OOB rows to a valid row: loads garbage, store is row-guarded, and
  // C rows depend only on their own A row. Keeps EXEC all-ones for WMMA.
  int arow_idx = m0 + l16;
  if (arow_idx >= M) arow_idx = M - 1;
  const unsigned short* arow = A + (size_t)arow_idx * K;
  const unsigned short* wrow = W + (size_t)(n0 + l16) * K;

  v8f acc[4] = {v8f{}, v8f{}, v8f{}, v8f{}};

  for (int k0 = 0; k0 < K; k0 += 32) {
    __builtin_prefetch(arow + k0 + 64, 0, 0);    // global_prefetch_b8
    v8u a  = frag_bf16(arow, k0, half);
    v8u b0 = frag_bf16(wrow,                     k0, half);
    v8u b1 = frag_bf16(wrow + (size_t)16 * K,    k0, half);
    v8u b2 = frag_bf16(wrow + (size_t)32 * K,    k0, half);
    v8u b3 = frag_bf16(wrow + (size_t)48 * K,    k0, half);
    acc[0] = WMMA_BF16(a, b0, acc[0]);
    acc[1] = WMMA_BF16(a, b1, acc[1]);
    acc[2] = WMMA_BF16(a, b2, acc[2]);
    acc[3] = WMMA_BF16(a, b3, acc[3]);
  }

  // C/D layout: VGPR r -> row m0 + half*8 + r, col n0 + t*16 + l16
#pragma unroll
  for (int t = 0; t < 4; ++t) {
    const int nn = n0 + t * 16 + l16;
#pragma unroll
    for (int r = 0; r < 8; ++r) {
      const int mm = m0 + half * 8 + r;
      if (mm < M) {
        float v = acc[t][r];
        if (BIAS) v += bias[nn];
        if (RES)  v += res[(size_t)mm * N + nn];
        if (RELU) v = fmaxf(v, 0.f);
        C[(size_t)mm * N + nn] = v;
      }
    }
  }
}

// ---------------------------------------------------------------------------
// message_atom[n] += sum_j(nei_j) * max_j(nei_j)   (elementwise, nei from a2b)
// ---------------------------------------------------------------------------
__global__ void agg_kernel(float* __restrict__ msg_atom,
                           const float* __restrict__ msg_bond,
                           const int* __restrict__ a2b, int N, int H, int MB) {
  const int per = H >> 2;
  int t = blockIdx.x * blockDim.x + threadIdx.x;
  if (t >= N * per) return;
  int n = t / per, c = (t - n * per) * 4;
  float s0 = 0, s1 = 0, s2 = 0, s3 = 0;
  float m0 = -INFINITY, m1 = -INFINITY, m2 = -INFINITY, m3 = -INFINITY;
  for (int j = 0; j < MB; ++j) {
    int e = a2b[n * MB + j];
    const float* p = msg_bond + (size_t)e * H + c;
    float v0 = p[0], v1 = p[1], v2 = p[2], v3 = p[3];
    s0 += v0; s1 += v1; s2 += v2; s3 += v3;
    m0 = fmaxf(m0, v0); m1 = fmaxf(m1, v1);
    m2 = fmaxf(m2, v2); m3 = fmaxf(m3, v3);
  }
  float* q = msg_atom + (size_t)n * H + c;
  q[0] += s0 * m0; q[1] += s1 * m1; q[2] += s2 * m2; q[3] += s3 * m3;
}

// mb[e] = message_atom[b2a[e]] - message_bond[b2revb[e]]
__global__ void edge_prep(float* __restrict__ mb,
                          const float* __restrict__ msg_atom,
                          const float* __restrict__ msg_bond,
                          const int* __restrict__ b2a,
                          const int* __restrict__ b2revb, int E, int H) {
  const int per = H >> 2;
  int t = blockIdx.x * blockDim.x + threadIdx.x;
  if (t >= E * per) return;
  int e = t / per, c = (t - e * per) * 4;
  const float* pa = msg_atom + (size_t)b2a[e] * H + c;
  const float* pb = msg_bond + (size_t)b2revb[e] * H + c;
  float* o = mb + (size_t)e * H + c;
  o[0] = pa[0] - pb[0]; o[1] = pa[1] - pb[1];
  o[2] = pa[2] - pb[2]; o[3] = pa[3] - pb[3];
}

// ---------------------------------------------------------------------------
// Attention: order-preserving float<->uint encode for atomic segment max
// ---------------------------------------------------------------------------
__device__ __forceinline__ unsigned f2ord(float f) {
  unsigned u = __builtin_bit_cast(unsigned, f);
  return (u & 0x80000000u) ? ~u : (u | 0x80000000u);
}
__device__ __forceinline__ float ord2f(unsigned k) {
  unsigned u = (k & 0x80000000u) ? (k & 0x7fffffffu) : ~k;
  return __builtin_bit_cast(float, u);
}

__global__ void attn_logits(const float* __restrict__ Q,
                            const float* __restrict__ Kmat,
                            const float* __restrict__ EAT,
                            const int* __restrict__ bonds, float* __restrict__ LOG,
                            unsigned* __restrict__ MX, int E, int H, int HEADS) {
  int t = blockIdx.x * blockDim.x + threadIdx.x;
  if (t >= E * HEADS) return;
  int e = t / HEADS, h = t - e * HEADS;
  int s = bonds[e], d = bonds[E + e];
  const int HDm = H * HEADS;
  const float4* q  = (const float4*)(Q    + (size_t)d * HDm + h * H);
  const float4* k  = (const float4*)(Kmat + (size_t)s * HDm + h * H);
  const float4* ea = (const float4*)(EAT  + (size_t)e * HDm + h * H);
  float acc = 0.f;
  for (int i = 0; i < (H >> 2); ++i) {
    float4 qq = q[i], kk = k[i], ee = ea[i];
    acc += qq.x * (kk.x + ee.x) + qq.y * (kk.y + ee.y) +
           qq.z * (kk.z + ee.z) + qq.w * (kk.w + ee.w);
  }
  float logit = acc * rsqrtf((float)H);
  LOG[t] = logit;
  atomicMax(&MX[(size_t)d * HEADS + h], f2ord(logit));
}

__global__ void attn_decode_max(unsigned* __restrict__ MX, int NH) {
  int t = blockIdx.x * blockDim.x + threadIdx.x;
  if (t >= NH) return;
  unsigned k = MX[t];
  float m = (k == 0u) ? 0.f : ord2f(k);        // empty segment -> 0 (ref)
  ((float*)MX)[t] = m;
}

__global__ void attn_exp(const int* __restrict__ bonds, float* __restrict__ LOG,
                         const float* __restrict__ MXF, float* __restrict__ DEN,
                         int E, int HEADS) {
  int t = blockIdx.x * blockDim.x + threadIdx.x;
  if (t >= E * HEADS) return;
  int e = t / HEADS, h = t - e * HEADS;
  int d = bonds[E + e];
  float ex = __expf(LOG[t] - MXF[(size_t)d * HEADS + h]);
  LOG[t] = ex;
  atomicAdd(&DEN[(size_t)d * HEADS + h], ex);
}

// out_mean[dst] += mean_h( (v[src]+eattr) * alpha )  via one atomic per (e,c)
__global__ void attn_scatter(const float* __restrict__ V,
                             const float* __restrict__ EAT,
                             const float* __restrict__ LOG,
                             const float* __restrict__ DEN,
                             const int* __restrict__ bonds,
                             float* __restrict__ OUT, int E, int H, int HEADS) {
  int t = blockIdx.x * blockDim.x + threadIdx.x;
  if (t >= E * H) return;
  int e = t / H, c = t - e * H;
  int s = bonds[e], d = bonds[E + e];
  const int HDm = H * HEADS;
  float acc = 0.f;
#pragma unroll
  for (int h = 0; h < 8; ++h) {   // HEADS == 8
    float alpha = LOG[(size_t)e * HEADS + h] /
                  (DEN[(size_t)d * HEADS + h] + 1e-16f);
    acc += (V[(size_t)s * HDm + h * H + c] + EAT[(size_t)e * HDm + h * H + c]) * alpha;
  }
  atomicAdd(&OUT[(size_t)d * H + c], acc * (1.f / (float)HEADS));
}

// ---------------------------------------------------------------------------
// Set2Set (T=1, zero init): qv is a constant vector from the LSTM biases.
// ---------------------------------------------------------------------------
__device__ __forceinline__ float sigmoidf(float x) { return 1.f / (1.f + __expf(-x)); }

__global__ void qv_kernel(const float* __restrict__ bih,
                          const float* __restrict__ bhh,
                          float* __restrict__ QV, int H) {
  int i = blockIdx.x * blockDim.x + threadIdx.x;
  if (i >= H) return;
  float gi = bih[i] + bhh[i];
  float gg = bih[2 * H + i] + bhh[2 * H + i];
  float go = bih[3 * H + i] + bhh[3 * H + i];
  float cc = sigmoidf(gi) * tanhf(gg);
  QV[i] = sigmoidf(go) * tanhf(cc);
}

// One wave per molecule: softmax-weighted readout r, writes QSTAR=[qv, r].
__global__ __launch_bounds__(128) void set2set_kernel(
    const float* __restrict__ HID, const float* __restrict__ QV,
    float* __restrict__ QSTAR, const int* __restrict__ a_size_p, int N, int H) {
  const int a_size = *a_size_p;
  const int n_mols = N / a_size;
  const int gwave = (int)((blockIdx.x * blockDim.x + threadIdx.x) >> 5);
  const int lane = threadIdx.x & 31;
  if (gwave >= n_mols) return;
  const float* x = HID + (size_t)gwave * a_size * H;
  const int nj = H >> 5;                    // H multiple of 32, <= 256
  float qv[8];
  for (int j = 0; j < nj; ++j) qv[j] = QV[lane + 32 * j];

  float mx = -INFINITY;
  for (int a = 0; a < a_size; ++a) {
    float p = 0.f;
    for (int j = 0; j < nj; ++j) p += x[(size_t)a * H + lane + 32 * j] * qv[j];
    for (int o = 16; o > 0; o >>= 1) p += __shfl_xor(p, o, 32);
    mx = fmaxf(mx, p);
  }
  float ssum = 0.f, r[8];
  for (int j = 0; j < nj; ++j) r[j] = 0.f;
  for (int a = 0; a < a_size; ++a) {
    float vals[8]; float p = 0.f;
    for (int j = 0; j < nj; ++j) {
      vals[j] = x[(size_t)a * H + lane + 32 * j];
      p += vals[j] * qv[j];
    }
    for (int o = 16; o > 0; o >>= 1) p += __shfl_xor(p, o, 32);
    float w = __expf(p - mx);
    ssum += w;
    for (int j = 0; j < nj; ++j) r[j] += w * vals[j];
  }
  float inv = 1.f / (ssum + 1e-16f);
  float* qs = QSTAR + (size_t)gwave * 2 * H;
  for (int j = 0; j < nj; ++j) {
    qs[lane + 32 * j]     = qv[j];
    qs[H + lane + 32 * j] = r[j] * inv;
  }
}

// ---------------------------------------------------------------------------
// Host launcher
// ---------------------------------------------------------------------------
extern "C" void kernel_launch(void* const* d_in, const int* in_sizes, int n_in,
                              void* d_out, int out_size, void* d_ws, size_t ws_size,
                              hipStream_t stream) {
  const float* f_atoms  = (const float*)d_in[0];
  const float* f_bonds  = (const float*)d_in[1];
  const float* wi_atom  = (const float*)d_in[2];
  const float* wi_bond  = (const float*)d_in[3];
  const float* wh[3]    = {(const float*)d_in[4], (const float*)d_in[5], (const float*)d_in[6]};
  const float* q_w      = (const float*)d_in[7];
  const float* q_b      = (const float*)d_in[8];
  const float* k_w      = (const float*)d_in[9];
  const float* k_b      = (const float*)d_in[10];
  const float* v_w      = (const float*)d_in[11];
  const float* v_b      = (const float*)d_in[12];
  const float* e_w      = (const float*)d_in[13];
  const float* skip_w   = (const float*)d_in[14];
  const float* skip_b   = (const float*)d_in[15];
  const float* lstm_bih = (const float*)d_in[18];
  const float* lstm_bhh = (const float*)d_in[19];
  const float* wo_w     = (const float*)d_in[20];
  const float* wo_b     = (const float*)d_in[21];
  const int*   a2b      = (const int*)d_in[22];
  const int*   b2a      = (const int*)d_in[23];
  const int*   b2revb   = (const int*)d_in[24];
  const int*   bonds    = (const int*)d_in[25];
  const int*   a_size   = (const int*)d_in[26];

  const int H  = in_sizes[15];            // 128
  const int FA = in_sizes[2] / H;         // 133
  const int FB = in_sizes[3] / H;         // 147
  const int Nn = in_sizes[0] / FA;        // 60000
  const int Ee = in_sizes[1] / FB;        // 120000
  const int HD = in_sizes[8];             // 1024
  const int HEADS = HD / H;               // 8
  const int MB = in_sizes[22] / Nn;       // 6
  const int FAp = (FA + 31) & ~31;        // 160
  const int FBp = (FB + 31) & ~31;        // 160
  const int H2  = 2 * H;                  // 256

  // Workspace carve-up (16B-aligned slices)
  float* w = (float*)d_ws;
  size_t o = 0;
  auto align4 = [&]() { o = (o + 3) & ~(size_t)3; };
  auto take  = [&](size_t n) { align4(); float* p = w + o; o += n; return p; };
  auto takeh = [&](size_t n_u16) {
    align4(); unsigned short* p = (unsigned short*)(w + o);
    o += (n_u16 + 1) / 2; return p;
  };
  // f32 buffers
  float* A0    = take((size_t)Nn * H);       // message_atom
  float* B0    = take((size_t)Ee * H);       // input_bond
  float* B1    = take((size_t)Ee * H);       // message_bond
  float* MBUF  = take((size_t)Ee * H);       // mb scratch
  float* Qm    = take((size_t)Nn * HD);
  float* Km    = take((size_t)Nn * HD);
  float* Vm    = take((size_t)Nn * HD);
  float* EAT   = take((size_t)Ee * HD);
  float* LOG   = take((size_t)Ee * HEADS);
  unsigned* MX = (unsigned*)take((size_t)Nn * HEADS);
  float* DEN   = take((size_t)Nn * HEADS);
  float* OUT   = take((size_t)Nn * H);
  float* HID   = take((size_t)Nn * H);
  float* QV    = take((size_t)H);
  float* QSTAR = take((size_t)Nn * H2);      // worst-case n_mols == N
  // bf16 buffers (u16 elements)
  unsigned short* fa_bf    = takeh((size_t)Nn * FAp);
  unsigned short* fb_bf    = takeh((size_t)Ee * FBp);
  unsigned short* wiA_bf   = takeh((size_t)H * FAp);
  unsigned short* wiB_bf   = takeh((size_t)H * FBp);
  unsigned short* wh_bf[3] = {takeh((size_t)H * H), takeh((size_t)H * H),
                              takeh((size_t)H * H)};
  unsigned short* qw_bf    = takeh((size_t)HD * H);
  unsigned short* kw_bf    = takeh((size_t)HD * H);
  unsigned short* vw_bf    = takeh((size_t)HD * H);
  unsigned short* ew_bf    = takeh((size_t)HD * H);
  unsigned short* skw_bf   = takeh((size_t)H * H);
  unsigned short* wow_bf   = takeh((size_t)H * H2);
  unsigned short* A0_bf    = takeh((size_t)Nn * H);
  unsigned short* B1_bf    = takeh((size_t)Ee * H);
  unsigned short* MB_bf    = takeh((size_t)Ee * H);
  unsigned short* QS_bf    = takeh((size_t)Nn * H2);

  const dim3 blk(128);
  auto ggrid = [](int M, int N) { return dim3((unsigned)((M + 63) / 64), (unsigned)(N / 64)); };
  auto cdiv  = [](long long a, int b) { return (unsigned)((a + b - 1) / b); };
  auto conv = [&](const float* s, unsigned short* d, int rows, int K, int Kp,
                  const int* rdiv) {
    to_bf16<<<cdiv((long long)rows * (Kp / 2), 256), 256, 0, stream>>>(
        s, (unsigned*)d, rows, K, Kp, rdiv);
  };

  // 0) convert all weights once
  conv(wi_atom, wiA_bf, H, FA, FAp, nullptr);
  conv(wi_bond, wiB_bf, H, FB, FBp, nullptr);
  for (int i = 0; i < 3; ++i) conv(wh[i], wh_bf[i], H, H, H, nullptr);
  conv(q_w, qw_bf, HD, H, H, nullptr);
  conv(k_w, kw_bf, HD, H, H, nullptr);
  conv(v_w, vw_bf, HD, H, H, nullptr);
  conv(e_w, ew_bf, HD, H, H, nullptr);
  conv(skip_w, skw_bf, H, H, H, nullptr);
  conv(wo_w, wow_bf, H, H2, H2, nullptr);

  // 1) input projections (+relu) on padded bf16 operands
  conv(f_atoms, fa_bf, Nn, FA, FAp, nullptr);
  conv(f_bonds, fb_bf, Ee, FB, FBp, nullptr);
  gemm_wmma<false, true, false><<<ggrid(Nn, H), blk, 0, stream>>>(
      fa_bf, wiA_bf, nullptr, nullptr, A0, Nn, H, FAp, nullptr);
  gemm_wmma<false, true, false><<<ggrid(Ee, H), blk, 0, stream>>>(
      fb_bf, wiB_bf, nullptr, nullptr, B0, Ee, H, FBp, nullptr);
  hipMemcpyAsync(B1, B0, (size_t)Ee * H * sizeof(float),
                 hipMemcpyDeviceToDevice, stream);   // message_bond = input_bond

  // 2) three message-passing iterations
  for (int it = 0; it < 3; ++it) {
    agg_kernel<<<cdiv((long long)Nn * (H / 4), 256), 256, 0, stream>>>(
        A0, B1, a2b, Nn, H, MB);
    edge_prep<<<cdiv((long long)Ee * (H / 4), 256), 256, 0, stream>>>(
        MBUF, A0, B1, b2a, b2revb, Ee, H);
    conv(MBUF, MB_bf, Ee, H, H, nullptr);
    gemm_wmma<false, true, true><<<ggrid(Ee, H), blk, 0, stream>>>(
        MB_bf, wh_bf[it], nullptr, B0, B1, Ee, H, H, nullptr);
  }

  // 3) TransformerConv projections
  conv(A0, A0_bf, Nn, H, H, nullptr);
  conv(B1, B1_bf, Ee, H, H, nullptr);
  gemm_wmma<true, false, false><<<ggrid(Nn, HD), blk, 0, stream>>>(
      A0_bf, qw_bf, q_b, nullptr, Qm, Nn, HD, H, nullptr);
  gemm_wmma<true, false, false><<<ggrid(Nn, HD), blk, 0, stream>>>(
      A0_bf, kw_bf, k_b, nullptr, Km, Nn, HD, H, nullptr);
  gemm_wmma<true, false, false><<<ggrid(Nn, HD), blk, 0, stream>>>(
      A0_bf, vw_bf, v_b, nullptr, Vm, Nn, HD, H, nullptr);
  gemm_wmma<false, false, false><<<ggrid(Ee, HD), blk, 0, stream>>>(
      B1_bf, ew_bf, nullptr, nullptr, EAT, Ee, HD, H, nullptr);

  // 4) segment softmax attention
  hipMemsetAsync(MX, 0, (size_t)Nn * HEADS * sizeof(unsigned), stream);
  hipMemsetAsync(DEN, 0, (size_t)Nn * HEADS * sizeof(float), stream);
  hipMemsetAsync(OUT, 0, (size_t)Nn * H * sizeof(float), stream);
  attn_logits<<<cdiv((long long)Ee * HEADS, 256), 256, 0, stream>>>(
      Qm, Km, EAT, bonds, LOG, MX, Ee, H, HEADS);
  attn_decode_max<<<cdiv((long long)Nn * HEADS, 256), 256, 0, stream>>>(
      MX, Nn * HEADS);
  attn_exp<<<cdiv((long long)Ee * HEADS, 256), 256, 0, stream>>>(
      bonds, LOG, (const float*)MX, DEN, Ee, HEADS);
  attn_scatter<<<cdiv((long long)Ee * H, 256), 256, 0, stream>>>(
      Vm, EAT, LOG, DEN, bonds, OUT, Ee, H, HEADS);

  // 5) atom_hiddens = out_mean + skip(message_atom)
  gemm_wmma<true, false, true><<<ggrid(Nn, H), blk, 0, stream>>>(
      A0_bf, skw_bf, skip_b, OUT, HID, Nn, H, H, nullptr);

  // 6) Set2Set (T=1) readout
  qv_kernel<<<cdiv(H, 128), 128, 0, stream>>>(lstm_bih, lstm_bhh, QV, H);
  set2set_kernel<<<cdiv(Nn, 4), 128, 0, stream>>>(HID, QV, QSTAR, a_size, Nn, H);

  // 7) final projection: out = q_star @ wo_w^T + wo_b   (M = Nn / *a_size)
  conv(QSTAR, QS_bf, Nn, H2, H2, a_size);
  gemm_wmma<true, false, false><<<ggrid(Nn, H), blk, 0, stream>>>(
      QS_bf, wow_bf, wo_b, nullptr, (float*)d_out, Nn, H, H2, a_size);
}
